// MoEPredictor_34376918238076
// MI455X (gfx1250) — matrix operand order repaired and checked by reading, not compiled, not run on hardware
//
#include <hip/hip_runtime.h>
#include <hip/hip_bf16.h>

// ---------------- constants -----------------
#define E_EXP 6
#define NTOK  (16384 * 6)          // B*M tokens
#define TT    32                   // tokens per workgroup
#define TRAJ_N 120                 // T*2
#define SCORES_OFF 11796480        // NTOK*TRAJ_N
#define AUX_OFF    (SCORES_OFF + NTOK)
#define PROBS_OFF  (AUX_OFF + 1)

// workspace offsets (in bf16/ushort units) for transposed-converted weights
#define OFF_RW1 0          // [256][128]
#define OFF_RW2 32768      // [128][256]
#define OFF_RW3 65536      // [16][128]      (6 valid cols, padded)
#define OFF_TW1 67584      // E x [256][128]
#define OFF_TW2 264192     // E x [256][256]
#define OFF_TW3 657408     // E x [128][256] (120 valid cols, padded)
#define OFF_SW1 854016     // E x [128][128]
#define OFF_SW2 952320     // E x [64][128]
#define OFF_SW3 1001472    // E x [16][64]   (1 valid col, padded)
#define GPSUM_BYTE_OFF 2015232     // = (1001472 + 6144) * 2

typedef __attribute__((ext_vector_type(16))) __bf16        v16bf;
typedef __attribute__((ext_vector_type(8)))  float         v8f;
typedef __attribute__((ext_vector_type(4)))  unsigned int  u32x4;

union Frag { v16bf v; u32x4 q[2]; };

__device__ __forceinline__ unsigned short f2bf(float f) {
  unsigned u = __builtin_bit_cast(unsigned, f);
  u += 0x7FFFu + ((u >> 16) & 1u);             // round-to-nearest-even
  return (unsigned short)(u >> 16);
}

// Branch-free erf (Abramowitz-Stegun 7.1.26, |err| < 1.5e-7), native rcp/exp.
__device__ __forceinline__ float erf_fast(float x) {
  const float ax = __builtin_fabsf(x);
  const float t  = __builtin_amdgcn_rcpf(__builtin_fmaf(0.3275911f, ax, 1.0f));
  float y = __builtin_fmaf(t, 1.061405429f, -1.453152027f);
  y = __builtin_fmaf(t, y, 1.421413741f);
  y = __builtin_fmaf(t, y, -0.284496736f);
  y = __builtin_fmaf(t, y, 0.254829592f);
  const float r = __builtin_fmaf(-y * t, __expf(-ax * ax), 1.0f);
  return __builtin_copysignf(r, x);
}
__device__ __forceinline__ float gelu_fast(float x) {
  return 0.5f * x * (1.0f + erf_fast(x * 0.70710678118654752440f));
}

// GEMM strip: [32 x kdim](LDS bf16) @ Wt[ncols][kdim](global bf16) + bias, GELU,
// write bf16 to LDS Out[32][ldo]. Each wave owns a 16-column strip and computes
// both 16-row tiles -> every B-fragment load feeds two WMMAs.
__device__ __forceinline__ void gemm_gelu(
    const unsigned short* __restrict__ A, int lda,
    const unsigned short* __restrict__ Wt,
    const float* __restrict__ bias,
    int kdim, int ncols,
    unsigned short* __restrict__ Out, int ldo,
    int wave, int lane)
{
  const int half = lane >> 4;       // 0/1
  const int ridx = lane & 15;
  const int strips = ncols >> 4;
  for (int s = wave; s < strips; s += 8) {
    const int n  = s * 16 + ridx;
    const float bv = bias[n];
    v8f acc0 = { bv, bv, bv, bv, bv, bv, bv, bv };
    v8f acc1 = acc0;
    const unsigned short* arow0 = A + ridx * lda;
    const unsigned short* arow1 = A + (16 + ridx) * lda;
    const unsigned short* brow  = Wt + (size_t)n * kdim;
    for (int k0 = 0; k0 < kdim; k0 += 32) {
      Frag a0, a1, b;
      const int ka = k0 + (half << 3);      // A: chunks at ka and ka+16
      const int kb = k0 + (half << 3);      // B: chunks at kb and kb+16
      b.q[0]  = *(const u32x4*)(brow + kb);
      b.q[1]  = *(const u32x4*)(brow + kb + 16);
      a0.q[0] = *(const u32x4*)(arow0 + ka);
      a0.q[1] = *(const u32x4*)(arow0 + ka + 16);
      a1.q[0] = *(const u32x4*)(arow1 + ka);
      a1.q[1] = *(const u32x4*)(arow1 + ka + 16);
      acc0 = __builtin_amdgcn_wmma_f32_16x16x32_bf16(
          false, a0.v, false, b.v, (short)0, acc0, false, false);
      acc1 = __builtin_amdgcn_wmma_f32_16x16x32_bf16(
          false, a1.v, false, b.v, (short)0, acc1, false, false);
    }
    const int mo = half << 3;
#pragma unroll
    for (int v = 0; v < 8; ++v) {
      Out[(mo + v) * ldo + n]      = f2bf(gelu_fast(acc0[v]));
      Out[(16 + mo + v) * ldo + n] = f2bf(gelu_fast(acc1[v]));
    }
  }
}

// Trajectory layer-3: [32x256] @ Wt[128][256] + bias (120 valid cols),
// weighted-accumulate into tacc[32][120] with sparse gate swv[t][e].
__device__ __forceinline__ void traj_l3(
    const unsigned short* __restrict__ A,
    const unsigned short* __restrict__ Wt,
    const float* __restrict__ bias,
    const float* __restrict__ swv, int e,
    float* __restrict__ tacc,
    int wave, int lane)
{
  const int half = lane >> 4;
  const int ridx = lane & 15;
  for (int s = wave; s < 8; s += 8) {     // 8 strips, 1 per wave
    const int n = s * 16 + ridx;
    const float bv = (n < TRAJ_N) ? bias[n] : 0.0f;
    v8f acc0 = { bv, bv, bv, bv, bv, bv, bv, bv };
    v8f acc1 = acc0;
    const unsigned short* arow0 = A + ridx * 256;
    const unsigned short* arow1 = A + (16 + ridx) * 256;
    const unsigned short* brow  = Wt + (size_t)n * 256;
    for (int k0 = 0; k0 < 256; k0 += 32) {
      Frag a0, a1, b;
      const int ka = k0 + (half << 3);
      const int kb = k0 + (half << 3);
      b.q[0]  = *(const u32x4*)(brow + kb);
      b.q[1]  = *(const u32x4*)(brow + kb + 16);
      a0.q[0] = *(const u32x4*)(arow0 + ka);
      a0.q[1] = *(const u32x4*)(arow0 + ka + 16);
      a1.q[0] = *(const u32x4*)(arow1 + ka);
      a1.q[1] = *(const u32x4*)(arow1 + ka + 16);
      acc0 = __builtin_amdgcn_wmma_f32_16x16x32_bf16(
          false, a0.v, false, b.v, (short)0, acc0, false, false);
      acc1 = __builtin_amdgcn_wmma_f32_16x16x32_bf16(
          false, a1.v, false, b.v, (short)0, acc1, false, false);
    }
    if (n < TRAJ_N) {
      const int mo = half << 3;
#pragma unroll
      for (int v = 0; v < 8; ++v) {
        const int r0 = mo + v;
        const int r1 = 16 + mo + v;
        tacc[r0 * TRAJ_N + n] += swv[r0 * E_EXP + e] * acc0[v];
        tacc[r1 * TRAJ_N + n] += swv[r1 * E_EXP + e] * acc1[v];
      }
    }
  }
}

// Router layer-3 as one padded WMMA strip on wave 0:
// [32x128](LDS) @ Wt[16][128] (6 valid cols) -> lgt[32][6] f32.
__device__ __forceinline__ void router_l3_wmma(
    const unsigned short* __restrict__ A,
    const unsigned short* __restrict__ Wt,
    const float* __restrict__ rb3,
    float* __restrict__ lgt,
    int wave, int lane)
{
  if (wave != 0) return;
  const int half = lane >> 4;
  const int ridx = lane & 15;
  const float bv = (ridx < E_EXP) ? rb3[ridx] : 0.0f;
  v8f acc0 = { bv, bv, bv, bv, bv, bv, bv, bv };
  v8f acc1 = acc0;
  const unsigned short* arow0 = A + ridx * 128;
  const unsigned short* arow1 = A + (16 + ridx) * 128;
  const unsigned short* brow  = Wt + ridx * 128;
  for (int k0 = 0; k0 < 128; k0 += 32) {
    Frag a0, a1, b;
    const int ka = k0 + (half << 3);
    const int kb = k0 + (half << 3);
    b.q[0]  = *(const u32x4*)(brow + kb);
    b.q[1]  = *(const u32x4*)(brow + kb + 16);
    a0.q[0] = *(const u32x4*)(arow0 + ka);
    a0.q[1] = *(const u32x4*)(arow0 + ka + 16);
    a1.q[0] = *(const u32x4*)(arow1 + ka);
    a1.q[1] = *(const u32x4*)(arow1 + ka + 16);
    acc0 = __builtin_amdgcn_wmma_f32_16x16x32_bf16(
        false, a0.v, false, b.v, (short)0, acc0, false, false);
    acc1 = __builtin_amdgcn_wmma_f32_16x16x32_bf16(
        false, a1.v, false, b.v, (short)0, acc1, false, false);
  }
  if (ridx < E_EXP) {
    const int mo = half << 3;
#pragma unroll
    for (int v = 0; v < 8; ++v) {
      lgt[(mo + v) * E_EXP + ridx]      = acc0[v];
      lgt[(16 + mo + v) * E_EXP + ridx] = acc1[v];
    }
  }
}

// Score layer-3 as one padded WMMA strip on wave 0:
// [32x64](LDS) @ Wt[16][64] (col 0 valid) -> sacc[r] += swv[r][e]*score.
// Lanes 0 (rows 0-7,16-23) and 16 (rows 8-15,24-31) cover all 32 rows.
__device__ __forceinline__ void score_l3_wmma(
    const unsigned short* __restrict__ A,
    const unsigned short* __restrict__ Wt,
    float sb3e,
    const float* __restrict__ swv, int e,
    float* __restrict__ sacc,
    int wave, int lane)
{
  if (wave != 0) return;
  const int half = lane >> 4;
  const int ridx = lane & 15;
  const float bv = (ridx == 0) ? sb3e : 0.0f;
  v8f acc0 = { bv, bv, bv, bv, bv, bv, bv, bv };
  v8f acc1 = acc0;
  const unsigned short* arow0 = A + ridx * 64;
  const unsigned short* arow1 = A + (16 + ridx) * 64;
  const unsigned short* brow  = Wt + ridx * 64;
  for (int k0 = 0; k0 < 64; k0 += 32) {
    Frag a0, a1, b;
    const int ka = k0 + (half << 3);
    const int kb = k0 + (half << 3);
    b.q[0]  = *(const u32x4*)(brow + kb);
    b.q[1]  = *(const u32x4*)(brow + kb + 16);
    a0.q[0] = *(const u32x4*)(arow0 + ka);
    a0.q[1] = *(const u32x4*)(arow0 + ka + 16);
    a1.q[0] = *(const u32x4*)(arow1 + ka);
    a1.q[1] = *(const u32x4*)(arow1 + ka + 16);
    acc0 = __builtin_amdgcn_wmma_f32_16x16x32_bf16(
        false, a0.v, false, b.v, (short)0, acc0, false, false);
    acc1 = __builtin_amdgcn_wmma_f32_16x16x32_bf16(
        false, a1.v, false, b.v, (short)0, acc1, false, false);
  }
  if (ridx == 0) {
    const int mo = half << 3;
#pragma unroll
    for (int v = 0; v < 8; ++v) {
      const int r0 = mo + v;
      const int r1 = 16 + mo + v;
      sacc[r0] += swv[r0 * E_EXP + e] * acc0[v];
      sacc[r1] += swv[r1 * E_EXP + e] * acc1[v];
    }
  }
}

// ---------------- prep: fp32 [batch][K][Nv] -> bf16 [batch][Np][K] ----------
__global__ void cvt_t(const float* __restrict__ src, unsigned short* __restrict__ dst,
                      int batch, int K, int Nv, int Np)
{
  const int total = batch * Np * K;
  for (int idx = blockIdx.x * blockDim.x + threadIdx.x; idx < total;
       idx += gridDim.x * blockDim.x) {
    const int e   = idx / (Np * K);
    const int rem = idx % (Np * K);
    const int n = rem / K;
    const int k = rem % K;
    const float v = (n < Nv) ? src[((size_t)e * K + k) * Nv + n] : 0.0f;
    dst[idx] = f2bf(v);
  }
}

__global__ void moe_zero(float* gpsum) {
  if (threadIdx.x < E_EXP) gpsum[threadIdx.x] = 0.0f;
}

__global__ void moe_finish(const float* __restrict__ gpsum, float* __restrict__ out) {
  if (threadIdx.x == 0) {
    float s = 0.0f;
    for (int e = 0; e < E_EXP; ++e) {
      const float a = gpsum[e] / (float)NTOK;
      s += a * a;
    }
    out[AUX_OFF] = (float)E_EXP * s;
  }
}

// ---------------- fused MoE kernel -----------------------------------------
__global__ void __launch_bounds__(256) moe_fused(
    const float* __restrict__ x,
    const float* __restrict__ rb1, const float* __restrict__ rb2, const float* __restrict__ rb3,
    const float* __restrict__ tb1, const float* __restrict__ tb2, const float* __restrict__ tb3,
    const float* __restrict__ sb1, const float* __restrict__ sb2, const float* __restrict__ sb3,
    const unsigned short* __restrict__ W,
    float* __restrict__ gpsum, float* __restrict__ out)
{
  __shared__ __align__(16) unsigned short xs[TT * 128];
  __shared__ __align__(16) unsigned short hA[TT * 256];
  __shared__ __align__(16) unsigned short hB[TT * 256];
  __shared__ float tacc[TT * TRAJ_N];
  __shared__ float swv[TT * E_EXP];
  __shared__ float lgt[TT * E_EXP];
  __shared__ float sacc[TT];
  __shared__ float psum[E_EXP];

  const int tid = threadIdx.x;
  const int wave = tid >> 5, lane = tid & 31;
  const int token0 = blockIdx.x * TT;

  for (int i = tid; i < TT * 128; i += 256) xs[i] = f2bf(x[(size_t)token0 * 128 + i]);
  for (int i = tid; i < TT * TRAJ_N; i += 256) tacc[i] = 0.0f;
  for (int i = tid; i < TT * E_EXP; i += 256) swv[i] = 0.0f;
  if (tid < TT) sacc[tid] = 0.0f;
  if (tid < E_EXP) psum[tid] = 0.0f;
  __syncthreads();

  // ---- router: 128 -> 256 -> 128 -> 6 ----
  gemm_gelu(xs, 128, W + OFF_RW1, rb1, 128, 256, hA, 256, wave, lane);
  __syncthreads();
  gemm_gelu(hA, 256, W + OFF_RW2, rb2, 256, 128, hB, 128, wave, lane);
  __syncthreads();
  router_l3_wmma(hB, W + OFF_RW3, rb3, lgt, wave, lane);
  __syncthreads();
  if (tid < TT) {
    float l[E_EXP];
    float mx = -1e30f;
    for (int e = 0; e < E_EXP; ++e) { l[e] = lgt[tid * E_EXP + e]; mx = fmaxf(mx, l[e]); }
    float p[E_EXP];
    float se = 0.0f;
    for (int e = 0; e < E_EXP; ++e) { p[e] = __expf(l[e] - mx); se += p[e]; }
    const float inv = __builtin_amdgcn_rcpf(se);
    const size_t pb = (size_t)PROBS_OFF + (size_t)(token0 + tid) * E_EXP;
    for (int e = 0; e < E_EXP; ++e) {
      p[e] *= inv;
      out[pb + e] = p[e];
      atomicAdd(&psum[e], p[e]);
    }
    // top-2 + renormalized softmax gate
    int i1 = 0;
    for (int e = 1; e < E_EXP; ++e) if (l[e] > l[i1]) i1 = e;
    int i2 = (i1 == 0) ? 1 : 0;
    for (int e = 0; e < E_EXP; ++e) if (e != i1 && l[e] > l[i2]) i2 = e;
    const float e2 = __expf(l[i2] - l[i1]);
    const float s2 = 1.0f + e2;
    const float invs2 = __builtin_amdgcn_rcpf(s2);
    swv[tid * E_EXP + i1] = invs2;
    swv[tid * E_EXP + i2] = e2 * invs2;
  }
  __syncthreads();
  if (tid < E_EXP) atomicAdd(&gpsum[tid], psum[tid]);

  // ---- all-expert heads ----
  for (int e = 0; e < E_EXP; ++e) {
    // trajectory head: 128 -> 256 -> 256 -> 120 (weighted accumulate)
    gemm_gelu(xs, 128, W + OFF_TW1 + (size_t)e * 256 * 128, tb1 + e * 256,
              128, 256, hA, 256, wave, lane);
    __syncthreads();
    gemm_gelu(hA, 256, W + OFF_TW2 + (size_t)e * 256 * 256, tb2 + e * 256,
              256, 256, hB, 256, wave, lane);
    __syncthreads();
    traj_l3(hB, W + OFF_TW3 + (size_t)e * 128 * 256, tb3 + e * TRAJ_N,
            swv, e, tacc, wave, lane);
    // score head: 128 -> 128 -> 64 -> 1
    gemm_gelu(xs, 128, W + OFF_SW1 + (size_t)e * 128 * 128, sb1 + e * 128,
              128, 128, hA, 128, wave, lane);
    __syncthreads();
    gemm_gelu(hA, 128, W + OFF_SW2 + (size_t)e * 64 * 128, sb2 + e * 64,
              128, 64, hB, 64, wave, lane);
    __syncthreads();
    score_l3_wmma(hB, W + OFF_SW3 + (size_t)e * 16 * 64, sb3[e],
                  swv, e, sacc, wave, lane);
    __syncthreads();
  }

  // ---- outputs ----
  for (int i = tid; i < TT * TRAJ_N; i += 256)
    out[(size_t)token0 * TRAJ_N + i] = tacc[i];
  if (tid < TT)
    out[(size_t)SCORES_OFF + token0 + tid] = sacc[tid];
}

// ---------------- launcher --------------------------------------------------
extern "C" void kernel_launch(void* const* d_in, const int* in_sizes, int n_in,
                              void* d_out, int out_size, void* d_ws, size_t ws_size,
                              hipStream_t stream) {
  const float* x   = (const float*)d_in[0];
  const float* rW1 = (const float*)d_in[1];
  const float* rb1 = (const float*)d_in[2];
  const float* rW2 = (const float*)d_in[3];
  const float* rb2 = (const float*)d_in[4];
  const float* rW3 = (const float*)d_in[5];
  const float* rb3 = (const float*)d_in[6];
  const float* tW1 = (const float*)d_in[7];
  const float* tb1 = (const float*)d_in[8];
  const float* tW2 = (const float*)d_in[9];
  const float* tb2 = (const float*)d_in[10];
  const float* tW3 = (const float*)d_in[11];
  const float* tb3 = (const float*)d_in[12];
  const float* sW1 = (const float*)d_in[13];
  const float* sb1 = (const float*)d_in[14];
  const float* sW2 = (const float*)d_in[15];
  const float* sb2 = (const float*)d_in[16];
  const float* sW3 = (const float*)d_in[17];
  const float* sb3 = (const float*)d_in[18];

  unsigned short* W = (unsigned short*)d_ws;
  float* gpsum = (float*)((char*)d_ws + GPSUM_BYTE_OFF);
  float* out = (float*)d_out;

  auto launch_cvt = [&](const float* src, size_t dstOff, int batch, int K, int Nv, int Np) {
    const int total = batch * Np * K;
    int blocks = (total + 255) / 256;
    if (blocks > 2048) blocks = 2048;
    cvt_t<<<dim3(blocks), dim3(256), 0, stream>>>(src, W + dstOff, batch, K, Nv, Np);
  };

  launch_cvt(rW1, OFF_RW1, 1, 128, 256, 256);
  launch_cvt(rW2, OFF_RW2, 1, 256, 128, 128);
  launch_cvt(rW3, OFF_RW3, 1, 128,   6,  16);
  launch_cvt(tW1, OFF_TW1, E_EXP, 128, 256, 256);
  launch_cvt(tW2, OFF_TW2, E_EXP, 256, 256, 256);
  launch_cvt(tW3, OFF_TW3, E_EXP, 256, 120, 128);
  launch_cvt(sW1, OFF_SW1, E_EXP, 128, 128, 128);
  launch_cvt(sW2, OFF_SW2, E_EXP, 128,  64,  64);
  launch_cvt(sW3, OFF_SW3, E_EXP,  64,   1,  16);

  moe_zero<<<dim3(1), dim3(32), 0, stream>>>(gpsum);

  moe_fused<<<dim3(NTOK / TT), dim3(256), 0, stream>>>(
      x, rb1, rb2, rb3, tb1, tb2, tb3, sb1, sb2, sb3, W, gpsum, out);

  moe_finish<<<dim3(1), dim3(1), 0, stream>>>(gpsum, out);
}
